// AlignSeg_FA_29205777613411
// MI455X (gfx1250) — matrix operand order repaired and checked by reading, not compiled, not run on hardware
//
#include <hip/hip_runtime.h>
#include <hip/hip_bf16.h>

typedef __attribute__((ext_vector_type(16))) __bf16 v16bf;
typedef __attribute__((ext_vector_type(8)))  float  v8f;
typedef __attribute__((ext_vector_type(4))) unsigned int tdm_u32x4;
typedef __attribute__((ext_vector_type(8))) int          tdm_i32x8;
typedef __attribute__((ext_vector_type(4))) int          tdm_i32x4;

#define WMMA_BF16(A, B, C) \
  __builtin_amdgcn_wmma_f32_16x16x32_bf16(false, (A), false, (B), (short)0, (C), false, false)

// ---------------------------------------------------------------------------
// TDM: 2D bf16 tile load Global -> LDS (CDNA5 ISA)
// group0: count=1 | lds_addr | global_addr(57b) | type=2
// group1: data_size=1(2B) | tensor_dim0/1 | tile_dim0/1 | tensor_dim0_stride
// 6-arg toolchain form: (g0, g1, g2, g3, g4/unused, cpol)
// ---------------------------------------------------------------------------
__device__ __forceinline__ void tdm_load_2d_bf16(unsigned lds_off,
                                                 const __bf16* gptr,
                                                 unsigned tensor_w, unsigned tensor_h,
                                                 unsigned tile_w, unsigned tile_h,
                                                 unsigned stride_elems) {
  unsigned long long ga = (unsigned long long)gptr;
  tdm_u32x4 g0;
  g0[0] = 1u;                                            // count=1, is_restore=0
  g0[1] = lds_off;                                       // lds_addr (bytes)
  g0[2] = (unsigned)(ga & 0xffffffffull);                // global_addr[31:0]
  g0[3] = (unsigned)((ga >> 32) & 0x1ffffffull) | (2u << 30);  // [56:32] | type=2
  tdm_i32x8 g1;
  g1[0] = (int)(1u << 16);                               // mask=0, data_size=1 (2B)
  g1[1] = (int)((tensor_w & 0xffffu) << 16);             // tensor_dim0[15:0]
  g1[2] = (int)(((tensor_w >> 16) & 0xffffu) | ((tensor_h & 0xffffu) << 16));
  g1[3] = (int)(((tensor_h >> 16) & 0xffffu) | ((tile_w & 0xffffu) << 16));
  g1[4] = (int)(tile_h & 0xffffu);                       // tile_dim1, tile_dim2=0
  g1[5] = (int)stride_elems;                             // tensor_dim0_stride[31:0]
  g1[6] = 0;
  g1[7] = 0;
  tdm_i32x4 z4 = {0, 0, 0, 0};
  tdm_i32x8 z8 = {0, 0, 0, 0, 0, 0, 0, 0};
  __builtin_amdgcn_tensor_load_to_lds(g0, g1, z4, z4, z8, 0);
}

__device__ __forceinline__ unsigned lds_offset_of(const void* p) {
  // flat LDS pointers carry the wave-relative LDS byte address in the low 32 bits
  return (unsigned)(unsigned long long)p;
}

// ---------------------------------------------------------------------------
// hi/lo bf16 split helpers ("bf16x3" f32 emulation)
// ---------------------------------------------------------------------------
__device__ __forceinline__ void split_bf16(float v, __bf16* hi, __bf16* lo) {
  __bf16 h = (__bf16)v;
  *hi = h;
  *lo = (__bf16)(v - (float)h);
}

// ---------------------------------------------------------------------------
// Kernel 1: 2x2 maxpool + bf16 hi/lo split  x1[8,512,128,128] -> x1p_hi/lo
// ---------------------------------------------------------------------------
__global__ void __launch_bounds__(256)
k_maxpool_split(const float* __restrict__ x1,
                __bf16* __restrict__ ph, __bf16* __restrict__ pl) {
  size_t i = (size_t)blockIdx.x * blockDim.x + threadIdx.x;   // 16,777,216 outputs
  int x = (int)(i & 63);
  int y = (int)((i >> 6) & 63);
  size_t nc = i >> 12;
  const float* src = x1 + (nc << 14) + (size_t)(2 * y) * 128 + 2 * x;
  float a = src[0], b = src[1], c = src[128], d = src[129];
  float m = fmaxf(fmaxf(a, b), fmaxf(c, d));
  split_bf16(m, &ph[i], &pl[i]);
}

// ---------------------------------------------------------------------------
// Kernel 2: split both 1x1-conv weight tensors to bf16 hi/lo
// ---------------------------------------------------------------------------
__global__ void __launch_bounds__(256)
k_split_w(const float* __restrict__ wa, const float* __restrict__ wb,
          __bf16* __restrict__ ah, __bf16* __restrict__ al,
          __bf16* __restrict__ bh, __bf16* __restrict__ bl) {
  int i = blockIdx.x * 256 + threadIdx.x;                // 0..262143
  if (i < 131072) split_bf16(wa[i], &ah[i], &al[i]);
  else            { int j = i - 131072; split_bf16(wb[j], &bh[j], &bl[j]); }
}

// ---------------------------------------------------------------------------
// Kernel 3: split x2 to bf16 hi/lo
// ---------------------------------------------------------------------------
__global__ void __launch_bounds__(256)
k_split_x2(const float* __restrict__ x2,
           __bf16* __restrict__ h, __bf16* __restrict__ l) {
  size_t i = (size_t)blockIdx.x * blockDim.x + threadIdx.x;   // 8,388,608
  split_bf16(x2[i], &h[i], &l[i]);
}

// ---------------------------------------------------------------------------
// A fragment: 8+8 contiguous bf16 -> v16bf, zero conversions
// lane<16: K {0..7,16..23}; lane>=16: K {8..15,24..31} (ISA 16-bit A 16x32)
// ---------------------------------------------------------------------------
union AFrag { uint4 q[2]; v16bf v; };
__device__ __forceinline__ v16bf load_a_bf(const __bf16* __restrict__ arow,
                                           int kb, int arun) {
  AFrag f;
  f.q[0] = *(const uint4*)(arow + kb + arun);
  f.q[1] = *(const uint4*)(arow + kb + arun + 16);
  return f.v;
}

// B fragment from LDS panel laid out [K][16] bf16
__device__ __forceinline__ v16bf lds_b_frag(const __bf16* pcol, int kb, int brun) {
  v16bf b;
#pragma unroll
  for (int e = 0; e < 16; ++e) b[e] = pcol[(kb + (brun >> 1) + (e & 7) + ((e >> 3) << 4)) * 16];
  return b;
}

// ---------------------------------------------------------------------------
// Kernel 4: GEMM1  x1c = w_change[256x512] * x1p + bias  (bf16x3, TDM B-panel)
// grid (256 p-tiles, 8 batch); 8 waves; wave w -> co tiles {32w, 32w+16}
// ---------------------------------------------------------------------------
__global__ void __launch_bounds__(256)
k_gemm1(const __bf16* __restrict__ ph, const __bf16* __restrict__ pl,
        const __bf16* __restrict__ wh, const __bf16* __restrict__ wl,
        const float* __restrict__ bias,
        __bf16* __restrict__ ch_out, __bf16* __restrict__ cl_out) {
  __shared__ __bf16 panH[512 * 16];
  __shared__ __bf16 panL[512 * 16];
  int wave = threadIdx.x >> 5;
  int lane = threadIdx.x & 31;
  int p0   = blockIdx.x * 16;
  int nbat = blockIdx.y;

  if (wave == 0) {
    size_t base = ((size_t)nbat << 21) + p0;             // n*512*4096 + p0
    tdm_load_2d_bf16(lds_offset_of(panH), ph + base, 4096, 512, 16, 512, 4096);
    tdm_load_2d_bf16(lds_offset_of(panL), pl + base, 4096, 512, 16, 512, 4096);
    __builtin_amdgcn_s_wait_tensorcnt((short)0);
  }
  __syncthreads();

  int m    = lane & 15;
  int half = lane >> 4;
  int arun = half * 8;
  int brun = half * 16;
  int co_base = wave * 32;
  const __bf16* arow0h = wh + (size_t)(co_base + m) * 512;
  const __bf16* arow0l = wl + (size_t)(co_base + m) * 512;
  const __bf16* arow1h = arow0h + 16 * 512;
  const __bf16* arow1l = arow0l + 16 * 512;
  const __bf16* bH = panH + m;
  const __bf16* bL = panL + m;

  v8f c0 = {}; v8f c1 = {};
  for (int kb = 0; kb < 512; kb += 32) {
    v16bf a0h = load_a_bf(arow0h, kb, arun);
    v16bf a0l = load_a_bf(arow0l, kb, arun);
    v16bf a1h = load_a_bf(arow1h, kb, arun);
    v16bf a1l = load_a_bf(arow1l, kb, arun);
    v16bf bh  = lds_b_frag(bH, kb, brun);
    v16bf bl  = lds_b_frag(bL, kb, brun);
    c0 = WMMA_BF16(a0h, bh, c0);
    c0 = WMMA_BF16(a0h, bl, c0);
    c0 = WMMA_BF16(a0l, bh, c0);
    c1 = WMMA_BF16(a1h, bh, c1);
    c1 = WMMA_BF16(a1h, bl, c1);
    c1 = WMMA_BF16(a1l, bh, c1);
  }

  int rowb = half * 8;
  size_t obase = ((size_t)nbat << 20) + p0 + m;          // n*256*4096
#pragma unroll
  for (int r = 0; r < 8; ++r) {
    int co = co_base + rowb + r;
    float v = c0[r] + bias[co];
    size_t off = obase + ((size_t)co << 12);
    split_bf16(v, &ch_out[off], &cl_out[off]);
  }
#pragma unroll
  for (int r = 0; r < 8; ++r) {
    int co = co_base + 16 + rowb + r;
    float v = c1[r] + bias[co];
    size_t off = obase + ((size_t)co << 12);
    split_bf16(v, &ch_out[off], &cl_out[off]);
  }
}

// ---------------------------------------------------------------------------
// Kernel 5: GEMM2 over virtual concat [x1c; x2] + BN + ReLU -> y (f32)
// TDM stacks the two 256-row panels into one contiguous 512-row LDS panel.
// ---------------------------------------------------------------------------
__global__ void __launch_bounds__(256)
k_gemm2(const __bf16* __restrict__ c1h, const __bf16* __restrict__ c1l,
        const __bf16* __restrict__ x2h, const __bf16* __restrict__ x2l,
        const __bf16* __restrict__ wh,  const __bf16* __restrict__ wl,
        const float* __restrict__ bn_g, const float* __restrict__ bn_b,
        const float* __restrict__ bn_m, const float* __restrict__ bn_v,
        float* __restrict__ yout) {
  __shared__ __bf16 panH[512 * 16];
  __shared__ __bf16 panL[512 * 16];
  int wave = threadIdx.x >> 5;
  int lane = threadIdx.x & 31;
  int p0   = blockIdx.x * 16;
  int nbat = blockIdx.y;

  if (wave == 0) {
    size_t base = ((size_t)nbat << 20) + p0;             // n*256*4096 + p0
    unsigned lh = lds_offset_of(panH), ll = lds_offset_of(panL);
    tdm_load_2d_bf16(lh,        c1h + base, 4096, 256, 16, 256, 4096);
    tdm_load_2d_bf16(lh + 8192, x2h + base, 4096, 256, 16, 256, 4096);
    tdm_load_2d_bf16(ll,        c1l + base, 4096, 256, 16, 256, 4096);
    tdm_load_2d_bf16(ll + 8192, x2l + base, 4096, 256, 16, 256, 4096);
    __builtin_amdgcn_s_wait_tensorcnt((short)0);
  }
  __syncthreads();

  int m    = lane & 15;
  int half = lane >> 4;
  int arun = half * 8;
  int brun = half * 16;
  int co_base = wave * 32;
  const __bf16* arow0h = wh + (size_t)(co_base + m) * 512;
  const __bf16* arow0l = wl + (size_t)(co_base + m) * 512;
  const __bf16* arow1h = arow0h + 16 * 512;
  const __bf16* arow1l = arow0l + 16 * 512;
  const __bf16* bH = panH + m;
  const __bf16* bL = panL + m;

  v8f c0 = {}; v8f c1 = {};
  for (int kb = 0; kb < 512; kb += 32) {
    v16bf a0h = load_a_bf(arow0h, kb, arun);
    v16bf a0l = load_a_bf(arow0l, kb, arun);
    v16bf a1h = load_a_bf(arow1h, kb, arun);
    v16bf a1l = load_a_bf(arow1l, kb, arun);
    v16bf bh  = lds_b_frag(bH, kb, brun);
    v16bf bl  = lds_b_frag(bL, kb, brun);
    c0 = WMMA_BF16(a0h, bh, c0);
    c0 = WMMA_BF16(a0h, bl, c0);
    c0 = WMMA_BF16(a0l, bh, c0);
    c1 = WMMA_BF16(a1h, bh, c1);
    c1 = WMMA_BF16(a1h, bl, c1);
    c1 = WMMA_BF16(a1l, bh, c1);
  }

  int rowb = half * 8;
  float* outp = yout + ((size_t)nbat << 20) + p0 + m;
#pragma unroll
  for (int r = 0; r < 8; ++r) {
    int co = co_base + rowb + r;
    float s = bn_g[co] * rsqrtf(bn_v[co] + 1e-5f);
    float v = (c0[r] - bn_m[co]) * s + bn_b[co];
    outp[(size_t)co << 12] = fmaxf(v, 0.0f);
  }
#pragma unroll
  for (int r = 0; r < 8; ++r) {
    int co = co_base + 16 + rowb + r;
    float s = bn_g[co] * rsqrtf(bn_v[co] + 1e-5f);
    float v = (c1[r] - bn_m[co]) * s + bn_b[co];
    outp[(size_t)co << 12] = fmaxf(v, 0.0f);
  }
}

// ---------------------------------------------------------------------------
// Kernel 6: 3x3 conv 256->4, pad 1:  flow = w_cc2 (*) y
// ---------------------------------------------------------------------------
__global__ void __launch_bounds__(256)
k_conv3(const float* __restrict__ y, const float* __restrict__ w2,
        float* __restrict__ flow) {
  __shared__ float rows[3][64];
  int t = threadIdx.x;
  int x = t & 63, f = t >> 6;
  int yy = blockIdx.x, n = blockIdx.y;
  const float* ybase = y + ((size_t)n << 20);
  float acc = 0.0f;
  for (int c = 0; c < 256; ++c) {
    const float* ch = ybase + ((size_t)c << 12);
    if (t < 192) {
      int r = t >> 6, xx = t & 63;
      int ry = yy - 1 + r;
      rows[r][xx] = (ry >= 0 && ry < 64) ? ch[(ry << 6) + xx] : 0.0f;
    }
    __syncthreads();
    const float* wp = w2 + ((size_t)f * 256 + c) * 9;
    float l0 = (x > 0)  ? rows[0][x - 1] : 0.0f;
    float r0 = (x < 63) ? rows[0][x + 1] : 0.0f;
    float l1 = (x > 0)  ? rows[1][x - 1] : 0.0f;
    float r1 = (x < 63) ? rows[1][x + 1] : 0.0f;
    float l2 = (x > 0)  ? rows[2][x - 1] : 0.0f;
    float r2 = (x < 63) ? rows[2][x + 1] : 0.0f;
    acc += wp[0] * l0 + wp[1] * rows[0][x] + wp[2] * r0
         + wp[3] * l1 + wp[4] * rows[1][x] + wp[5] * r1
         + wp[6] * l2 + wp[7] * rows[2][x] + wp[8] * r2;
    __syncthreads();
  }
  flow[(((size_t)n * 4 + f) << 12) + (yy << 6) + x] = acc;
}

// ---------------------------------------------------------------------------
// Kernel 7: dual bilinear grid-sample (align_corners=True, zero pad) + add
// x1c reconstructed from bf16 hi+lo; x2 sampled from original f32 input
// ---------------------------------------------------------------------------
__device__ __forceinline__ float gs_fetch_f32(const float* __restrict__ img,
                                              int x, int y) {
  if (x < 0 || x > 63 || y < 0 || y > 63) return 0.0f;
  return img[(y << 6) + x];
}
__device__ __forceinline__ float gs_fetch_split(const __bf16* __restrict__ hi,
                                                const __bf16* __restrict__ lo,
                                                int x, int y) {
  if (x < 0 || x > 63 || y < 0 || y > 63) return 0.0f;
  int i = (y << 6) + x;
  return (float)hi[i] + (float)lo[i];
}

__global__ void __launch_bounds__(256)
k_sample(const __bf16* __restrict__ c1h, const __bf16* __restrict__ c1l,
         const float* __restrict__ x2, const float* __restrict__ flow,
         float* __restrict__ out) {
  int lane = threadIdx.x & 31;
  int ty   = threadIdx.x >> 5;
  int xch = blockIdx.x, yy = blockIdx.y;
  int n = blockIdx.z >> 5, cg = blockIdx.z & 31;
  int x = xch * 32 + lane;
  int c = cg * 8 + ty;
  size_t pix = ((size_t)yy << 6) + x;
  const float* fl = flow + (((size_t)n * 4) << 12) + pix;
  float ix1 = (float)x + fl[0 << 12];
  float iy1 = (float)yy + fl[1 << 12];
  float ix2 = (float)x + fl[2 << 12];
  float iy2 = (float)yy + fl[3 << 12];
  size_t chan = ((size_t)n * 256 + c) << 12;

  float x0f = floorf(ix1), y0f = floorf(iy1);
  int x0 = (int)x0f, y0 = (int)y0f;
  float tx = ix1 - x0f, tyw = iy1 - y0f;
  const __bf16* h = c1h + chan;
  const __bf16* l = c1l + chan;
  float u1 = gs_fetch_split(h, l, x0, y0)         * (1.0f - tx) * (1.0f - tyw)
           + gs_fetch_split(h, l, x0 + 1, y0)     * tx * (1.0f - tyw)
           + gs_fetch_split(h, l, x0, y0 + 1)     * (1.0f - tx) * tyw
           + gs_fetch_split(h, l, x0 + 1, y0 + 1) * tx * tyw;

  float x1f = floorf(ix2), y1f = floorf(iy2);
  int x1i = (int)x1f, y1i = (int)y1f;
  float tx2 = ix2 - x1f, ty2 = iy2 - y1f;
  const float* img2 = x2 + chan;
  float u2 = gs_fetch_f32(img2, x1i, y1i)         * (1.0f - tx2) * (1.0f - ty2)
           + gs_fetch_f32(img2, x1i + 1, y1i)     * tx2 * (1.0f - ty2)
           + gs_fetch_f32(img2, x1i, y1i + 1)     * (1.0f - tx2) * ty2
           + gs_fetch_f32(img2, x1i + 1, y1i + 1) * tx2 * ty2;

  out[chan + pix] = u1 + u2;
}

// ---------------------------------------------------------------------------
// ws layout (bytes), lifetime-overlaid, peak ~97 MB:
//   [0,        33.5M)  x1p_hi   -> later x2_hi/x2_lo -> later flow
//   [33.5M,    67.1M)  x1p_lo   -> later y (f32)
//   [67.1M,    68.2M)  wch_hi, wch_lo, wc1_hi, wc1_lo (256 KB each)
//   [68.2M,    84.9M)  x1c_hi
//   [84.9M,   101.7M)  x1c_lo
// ---------------------------------------------------------------------------
extern "C" void kernel_launch(void* const* d_in, const int* in_sizes, int n_in,
                              void* d_out, int out_size, void* d_ws, size_t ws_size,
                              hipStream_t stream) {
  const float* x1       = (const float*)d_in[0];
  const float* x2       = (const float*)d_in[1];
  const float* w_change = (const float*)d_in[2];
  const float* b_change = (const float*)d_in[3];
  const float* w_cc1    = (const float*)d_in[4];
  const float* bn_g     = (const float*)d_in[5];
  const float* bn_b     = (const float*)d_in[6];
  const float* bn_m     = (const float*)d_in[7];
  const float* bn_v     = (const float*)d_in[8];
  const float* w_cc2    = (const float*)d_in[9];
  float* out = (float*)d_out;
  char*  wsb = (char*)d_ws;

  __bf16* x1p_hi = (__bf16*)(wsb + 0);
  __bf16* x1p_lo = (__bf16*)(wsb + 33554432);
  __bf16* x2_hi  = (__bf16*)(wsb + 0);            // overlays dead x1p_hi
  __bf16* x2_lo  = (__bf16*)(wsb + 16777216);
  float*  yact   = (float*) (wsb + 33554432);     // overlays dead x1p_lo
  float*  flow   = (float*) (wsb + 0);            // overlays dead x2 splits
  __bf16* wch_hi = (__bf16*)(wsb + 67108864);
  __bf16* wch_lo = (__bf16*)(wsb + 67371008);
  __bf16* wc1_hi = (__bf16*)(wsb + 67633152);
  __bf16* wc1_lo = (__bf16*)(wsb + 67895296);
  __bf16* x1c_hi = (__bf16*)(wsb + 68157440);
  __bf16* x1c_lo = (__bf16*)(wsb + 84934656);

  k_maxpool_split<<<65536, 256, 0, stream>>>(x1, x1p_hi, x1p_lo);
  k_split_w<<<1024, 256, 0, stream>>>(w_change, w_cc1, wch_hi, wch_lo, wc1_hi, wc1_lo);
  k_gemm1<<<dim3(256, 8), 256, 0, stream>>>(x1p_hi, x1p_lo, wch_hi, wch_lo,
                                            b_change, x1c_hi, x1c_lo);
  k_split_x2<<<32768, 256, 0, stream>>>(x2, x2_hi, x2_lo);
  k_gemm2<<<dim3(256, 8), 256, 0, stream>>>(x1c_hi, x1c_lo, x2_hi, x2_lo,
                                            wc1_hi, wc1_lo, bn_g, bn_b, bn_m, bn_v, yact);
  k_conv3<<<dim3(64, 8), 256, 0, stream>>>(yact, w_cc2, flow);
  k_sample<<<dim3(2, 64, 256), 256, 0, stream>>>(x1c_hi, x1c_lo, x2, flow, out);
}
